// MultiHeadSelfAttention_67095979099121
// MI455X (gfx1250) — matrix operand order repaired and checked by reading, not compiled, not run on hardware
//
#include <hip/hip_runtime.h>

// MI455X / gfx1250, wave32, WMMA 16x16x32 bf16 with fp32 accumulation.
// MHSA: B=2, S=2048, D=1024, H=16, Dh=64, causal, fp32 I/O.
// v2: double-buffered LDS staging in both GEMMs (overlap HBM latency with
// WMMA), software-pipelined B-fragment LDS loads (partial dscnt waits),
// global_prefetch of next K/V block in flash attention.

#define D_MODEL 1024
#define NHEAD   16
#define DHEAD   64
#define BATCH   2
#define SEQ     2048

typedef __attribute__((ext_vector_type(8)))  float  v8f;
typedef __attribute__((ext_vector_type(4)))  float  v4f;
typedef __attribute__((ext_vector_type(16))) __bf16 v16bf;
typedef __attribute__((ext_vector_type(8)))  __bf16 v8bf;

#define WMMA_BF16(a, b, c) \
  __builtin_amdgcn_wmma_f32_16x16x32_bf16(false, (a), false, (b), (short)0, (c), false, false)

// A-fragment 16(M)x32(K) bf16.  ISA 7.12.2: lanes 0-15 hold M=0..15 with
// K=0..7 / 16..23; lanes 16-31 same rows with K=8..15 / 24..31.
__device__ __forceinline__ v16bf frag_a(const __bf16* base, int ld) {
  const int lane = threadIdx.x & 31;
  const __bf16* p = base + (lane & 15) * ld + ((lane >> 4) << 3);
  v8bf lo = *(const v8bf*)(p);
  v8bf hi = *(const v8bf*)(p + 16);
  v16bf f;
#pragma unroll
  for (int i = 0; i < 8; ++i) { f[i] = lo[i]; f[i + 8] = hi[i]; }
  return f;
}

// B-fragment 32(K)x16(N) bf16: lanes 0-15 = column N with K=0..7 / 16..23,
// lanes 16-31 K=8..15 / 24..31.  Column n of B at base + n*ld, K contiguous.
__device__ __forceinline__ v16bf frag_b(const __bf16* base, int ld) {
  const int lane = threadIdx.x & 31;
  const __bf16* p = base + (lane & 15) * ld + ((lane >> 4) << 3);
  v8bf lo = *(const v8bf*)(p);
  v8bf hi = *(const v8bf*)(p + 16);
  v16bf f;
#pragma unroll
  for (int i = 0; i < 8; ++i) { f[i] = lo[i]; f[i + 8] = hi[i]; }
  return f;
}

// Convert 16 staged fp32 values -> bf16 row chunk in LDS (compiler packs
// into v_cvt_pk_bf16_f32 + ds_store_b128).
__device__ __forceinline__ void stage16(__bf16* dst, const v4f* r) {
#pragma unroll
  for (int i = 0; i < 4; ++i) {
    dst[i * 4 + 0] = (__bf16)r[i].x; dst[i * 4 + 1] = (__bf16)r[i].y;
    dst[i * 4 + 2] = (__bf16)r[i].z; dst[i * 4 + 3] = (__bf16)r[i].w;
  }
}

// ---------------------------------------------------------------------------
// Kernel 1: fused QKV projection.  y = x W^T, 128x128 tile per workgroup,
// blockIdx.z = 0/1/2 -> Q/K/V.  Q,K written [b][h][s][d]; V written [b][h][d][s].
// ---------------------------------------------------------------------------
__global__ __launch_bounds__(256) void qkv_proj_kernel(
    const float* __restrict__ x,  const float* __restrict__ wq,
    const float* __restrict__ wk, const float* __restrict__ wv,
    __bf16* __restrict__ Qh, __bf16* __restrict__ Kh, __bf16* __restrict__ Vt) {
  __shared__ __bf16 sA[2][128 * 40];
  __shared__ __bf16 sB[2][128 * 40];

  const int mode = blockIdx.z;
  const float* W = (mode == 0) ? wq : (mode == 1) ? wk : wv;
  const int m0 = blockIdx.x * 128;
  const int n0 = blockIdx.y * 128;
  const int wave = threadIdx.x >> 5;
  const int lane = threadIdx.x & 31;

  v8f acc[8] = {};

  const int ldr = threadIdx.x >> 1;        // 0..127 (row within tile)
  const int ldc = (threadIdx.x & 1) * 16;  // 0 or 16 (k offset)
  const float* srcA = x + (size_t)(m0 + ldr) * D_MODEL + ldc;
  const float* srcW = W + (size_t)(n0 + ldr) * D_MODEL + ldc;
  __bf16* dstA = &sA[0][ldr * 40 + ldc];   // buffer selected via +offset
  __bf16* dstB = &sB[0][ldr * 40 + ldc];
  const int bufStride = 128 * 40;

  // Prologue: stage k-tile 0 into buffer 0.
  v4f ra[4], rw[4];
#pragma unroll
  for (int i = 0; i < 4; ++i) {
    ra[i] = *(const v4f*)(srcA + i * 4);
    rw[i] = *(const v4f*)(srcW + i * 4);
  }
  stage16(dstA, ra);
  stage16(dstB, rw);

  for (int kt = 0; kt < D_MODEL / 32; ++kt) {
    const int cur = kt & 1;
    const bool more = (kt + 1) < (D_MODEL / 32);
    if (more) {  // issue next tile's global loads before waiting/computing
      const float* a2 = srcA + (kt + 1) * 32;
      const float* w2 = srcW + (kt + 1) * 32;
#pragma unroll
      for (int i = 0; i < 4; ++i) {
        ra[i] = *(const v4f*)(a2 + i * 4);
        rw[i] = *(const v4f*)(w2 + i * 4);
      }
    }
    __syncthreads();  // buffer `cur` fully staged by all waves

    const __bf16* bA = &sA[0][cur * bufStride];
    const __bf16* bB = &sB[0][cur * bufStride];
    v16bf af = frag_a(bA + wave * 16 * 40, 40);
    v16bf bc = frag_b(bB, 40);
#pragma unroll
    for (int nt = 0; nt < 8; ++nt) {  // pipeline B-frag loads past WMMA
      v16bf bn = (nt < 7) ? frag_b(bB + (nt + 1) * 16 * 40, 40) : bc;
      acc[nt] = WMMA_BF16(af, bc, acc[nt]);
      bc = bn;
    }

    if (more) {  // stage next tile into the other buffer (no barrier needed:
                 // all waves are between the two syncthreads, buffer cur^1
                 // is write-only in this window)
      stage16(dstA + (cur ^ 1) * bufStride, ra);
      stage16(dstB + (cur ^ 1) * bufStride, rw);
    }
  }

  // Epilogue: scatter into head-major layouts; fold 1/sqrt(Dh) into Q.
  const int hi  = lane >> 4;
  const int col = lane & 15;
  const float qscale = (mode == 0) ? 0.125f : 1.0f;
#pragma unroll
  for (int nt = 0; nt < 8; ++nt) {
#pragma unroll
    for (int r = 0; r < 8; ++r) {
      const int m = m0 + wave * 16 + r + hi * 8;   // global row (b*S+s)
      const int e = n0 + nt * 16 + col;            // global output col
      const int b = m >> 11, s = m & 2047;
      const int h = e >> 6,  d = e & 63;
      const __bf16 bv = (__bf16)(acc[nt][r] * qscale);
      if (mode == 0)
        Qh[(((size_t)b * NHEAD + h) * SEQ + s) * DHEAD + d] = bv;
      else if (mode == 1)
        Kh[(((size_t)b * NHEAD + h) * SEQ + s) * DHEAD + d] = bv;
      else
        Vt[(((size_t)b * NHEAD + h) * DHEAD + d) * SEQ + s] = bv;
    }
  }
}

// ---------------------------------------------------------------------------
// Kernel 2: causal flash attention.  One wave owns a 16-row q tile; 32 keys
// per iteration (2 S-tiles of 4 WMMA), online softmax in fp32 C-layout regs,
// P converted C-layout -> A-fragment via per-wave LDS, 4 WMMA for P*V^T.
// ---------------------------------------------------------------------------
__global__ __launch_bounds__(256) void flash_attn_kernel(
    const __bf16* __restrict__ Qh, const __bf16* __restrict__ Kh,
    const __bf16* __restrict__ Vt, __bf16* __restrict__ Obuf) {
  __shared__ __bf16 sP[8 * 16 * 40];

  const int wave = threadIdx.x >> 5;
  const int lane = threadIdx.x & 31;
  const int hi   = lane >> 4;
  const int col  = lane & 15;
  const int bh = blockIdx.y;          // b*16 + h
  const int b = bh >> 4, h = bh & 15;
  const int q0 = blockIdx.x * 128 + wave * 16;

  const __bf16* Qbase = Qh + ((size_t)bh * SEQ + q0) * DHEAD;
  const v16bf qa0 = frag_a(Qbase, DHEAD);        // d = 0..31
  const v16bf qa1 = frag_a(Qbase + 32, DHEAD);   // d = 32..63

  v8f o0 = {}, o1 = {}, o2 = {}, o3 = {};
  float mrow[8], lrow[8];
#pragma unroll
  for (int r = 0; r < 8; ++r) { mrow[r] = -1e30f; lrow[r] = 0.0f; }

  __bf16* pP = sP + wave * 16 * 40;
  const __bf16* Krows = Kh + (size_t)bh * SEQ * DHEAD;   // [s][d]
  const __bf16* Vrows = Vt + (size_t)bh * DHEAD * SEQ;   // [d][s]
  const int jmax = (q0 + 15) >> 5;   // last 32-key block touching the diagonal

  for (int j = 0; j <= jmax; ++j) {
    const int k0 = j * 32;

    if (j < jmax) {  // prefetch next block's K rows and V columns
      __builtin_prefetch(Krows + (size_t)(k0 + 32 + lane) * DHEAD, 0, 3);
      __builtin_prefetch(Vrows + (size_t)lane * SEQ + k0 + 32, 0, 3);
      __builtin_prefetch(Vrows + (size_t)(lane + 32) * SEQ + k0 + 32, 0, 3);
    }

    const __bf16* Kbase = Krows + (size_t)k0 * DHEAD;
    v8f s0 = {}, s1 = {};
    s0 = WMMA_BF16(qa0, frag_b(Kbase, DHEAD), s0);
    s0 = WMMA_BF16(qa1, frag_b(Kbase + 32, DHEAD), s0);
    s1 = WMMA_BF16(qa0, frag_b(Kbase + 16 * DHEAD, DHEAD), s1);
    s1 = WMMA_BF16(qa1, frag_b(Kbase + 16 * DHEAD + 32, DHEAD), s1);

    if (k0 + 31 > q0) {  // diagonal block: apply causal mask
#pragma unroll
      for (int r = 0; r < 8; ++r) {
        const int q = q0 + r + hi * 8;
        if (k0 + col > q)      s0[r] = -1e30f;
        if (k0 + 16 + col > q) s1[r] = -1e30f;
      }
    }

#pragma unroll
    for (int r = 0; r < 8; ++r) {
      // row max across the 32 keys (16 lanes per row half, wave32)
      float mb = fmaxf(s0[r], s1[r]);
      mb = fmaxf(mb, __shfl_xor(mb, 1));
      mb = fmaxf(mb, __shfl_xor(mb, 2));
      mb = fmaxf(mb, __shfl_xor(mb, 4));
      mb = fmaxf(mb, __shfl_xor(mb, 8));
      const float mn    = fmaxf(mrow[r], mb);
      const float alpha = __expf(mrow[r] - mn);
      mrow[r] = mn;
      const float p0 = __expf(s0[r] - mn);
      const float p1 = __expf(s1[r] - mn);
      float rs = p0 + p1;
      rs += __shfl_xor(rs, 1); rs += __shfl_xor(rs, 2);
      rs += __shfl_xor(rs, 4); rs += __shfl_xor(rs, 8);
      lrow[r] = lrow[r] * alpha + rs;
      o0[r] *= alpha; o1[r] *= alpha; o2[r] *= alpha; o3[r] *= alpha;
      // C-layout -> LDS (row-major 16x32, stride 40).  DS ops are in-order
      // within a wave, so the frag_a readback below is safe without a barrier.
      pP[(r + hi * 8) * 40 + col]      = (__bf16)p0;
      pP[(r + hi * 8) * 40 + col + 16] = (__bf16)p1;
    }

    const v16bf pa = frag_a(pP, 40);
    const __bf16* Vbase = Vrows + k0;
    o0 = WMMA_BF16(pa, frag_b(Vbase, SEQ), o0);
    o1 = WMMA_BF16(pa, frag_b(Vbase + 16 * SEQ, SEQ), o1);
    o2 = WMMA_BF16(pa, frag_b(Vbase + 32 * SEQ, SEQ), o2);
    o3 = WMMA_BF16(pa, frag_b(Vbase + 48 * SEQ, SEQ), o3);
  }

  // Epilogue: O /= l, write bf16 into [b][s][h*64+d] for the output GEMM.
  __bf16* Od = Obuf + (size_t)b * SEQ * D_MODEL + (size_t)h * DHEAD;
#pragma unroll
  for (int r = 0; r < 8; ++r) {
    const float inv = 1.0f / lrow[r];
    const int q = q0 + r + hi * 8;
    __bf16* row = Od + (size_t)q * D_MODEL + col;
    row[0]  = (__bf16)(o0[r] * inv);
    row[16] = (__bf16)(o1[r] * inv);
    row[32] = (__bf16)(o2[r] * inv);
    row[48] = (__bf16)(o3[r] * inv);
  }
}

// ---------------------------------------------------------------------------
// Kernel 3: output projection.  out = O W_o^T, bf16 A, fp32 weights staged
// to bf16, fp32 output.  Same double-buffered pipeline as kernel 1.
// ---------------------------------------------------------------------------
__global__ __launch_bounds__(256) void out_proj_kernel(
    const __bf16* __restrict__ A, const float* __restrict__ Wo,
    float* __restrict__ out) {
  __shared__ __bf16 sA[2][128 * 40];
  __shared__ __bf16 sB[2][128 * 40];

  const int m0 = blockIdx.x * 128;
  const int n0 = blockIdx.y * 128;
  const int wave = threadIdx.x >> 5;
  const int lane = threadIdx.x & 31;

  v8f acc[8] = {};

  const int ldr = threadIdx.x >> 1;
  const int ldc = (threadIdx.x & 1) * 16;
  const __bf16* srcA = A + (size_t)(m0 + ldr) * D_MODEL + ldc;
  const float*  srcW = Wo + (size_t)(n0 + ldr) * D_MODEL + ldc;
  __bf16* dstA = &sA[0][ldr * 40 + ldc];
  __bf16* dstB = &sB[0][ldr * 40 + ldc];
  const int bufStride = 128 * 40;

  v8bf ra0, ra1;
  v4f rw[4];
  ra0 = *(const v8bf*)(srcA);
  ra1 = *(const v8bf*)(srcA + 8);
#pragma unroll
  for (int i = 0; i < 4; ++i) rw[i] = *(const v4f*)(srcW + i * 4);
  *(v8bf*)(dstA)     = ra0;
  *(v8bf*)(dstA + 8) = ra1;
  stage16(dstB, rw);

  for (int kt = 0; kt < D_MODEL / 32; ++kt) {
    const int cur = kt & 1;
    const bool more = (kt + 1) < (D_MODEL / 32);
    if (more) {
      const __bf16* a2 = srcA + (kt + 1) * 32;
      const float*  w2 = srcW + (kt + 1) * 32;
      ra0 = *(const v8bf*)(a2);
      ra1 = *(const v8bf*)(a2 + 8);
#pragma unroll
      for (int i = 0; i < 4; ++i) rw[i] = *(const v4f*)(w2 + i * 4);
    }
    __syncthreads();

    const __bf16* bA = &sA[0][cur * bufStride];
    const __bf16* bB = &sB[0][cur * bufStride];
    v16bf af = frag_a(bA + wave * 16 * 40, 40);
    v16bf bc = frag_b(bB, 40);
#pragma unroll
    for (int nt = 0; nt < 8; ++nt) {
      v16bf bn = (nt < 7) ? frag_b(bB + (nt + 1) * 16 * 40, 40) : bc;
      acc[nt] = WMMA_BF16(af, bc, acc[nt]);
      bc = bn;
    }

    if (more) {
      __bf16* da = dstA + (cur ^ 1) * bufStride;
      *(v8bf*)(da)     = ra0;
      *(v8bf*)(da + 8) = ra1;
      stage16(dstB + (cur ^ 1) * bufStride, rw);
    }
  }

  const int hi  = lane >> 4;
  const int col = lane & 15;
#pragma unroll
  for (int nt = 0; nt < 8; ++nt) {
#pragma unroll
    for (int r = 0; r < 8; ++r) {
      const int m = m0 + wave * 16 + r + hi * 8;
      const int e = n0 + nt * 16 + col;
      out[(size_t)m * D_MODEL + e] = acc[nt][r];
    }
  }
}

// ---------------------------------------------------------------------------
extern "C" void kernel_launch(void* const* d_in, const int* in_sizes, int n_in,
                              void* d_out, int out_size, void* d_ws, size_t ws_size,
                              hipStream_t stream) {
  const float* x  = (const float*)d_in[0];
  const float* wq = (const float*)d_in[1];
  const float* wk = (const float*)d_in[2];
  const float* wv = (const float*)d_in[3];
  const float* wo = (const float*)d_in[4];
  float* out = (float*)d_out;

  // Workspace: 4 x 8MB bf16 buffers (Q, K, V^T, attention output) = 32 MB.
  char* ws = (char*)d_ws;
  __bf16* Qh = (__bf16*)(ws);
  __bf16* Kh = (__bf16*)(ws + ((size_t)8  << 20));
  __bf16* Vt = (__bf16*)(ws + ((size_t)16 << 20));
  __bf16* Ob = (__bf16*)(ws + ((size_t)24 << 20));

  // 1) Q/K/V projections: grid (M/128, N/128, 3), 256 threads (8 waves).
  qkv_proj_kernel<<<dim3((BATCH * SEQ) / 128, D_MODEL / 128, 3), 256, 0, stream>>>(
      x, wq, wk, wv, Qh, Kh, Vt);

  // 2) Causal flash attention: grid (S/128, B*H), one wave per 16 q-rows.
  flash_attn_kernel<<<dim3(SEQ / 128, BATCH * NHEAD), 256, 0, stream>>>(
      Qh, Kh, Vt, Ob);

  // 3) Output projection.
  out_proj_kernel<<<dim3((BATCH * SEQ) / 128, D_MODEL / 128), 256, 0, stream>>>(
      Ob, wo, out);
}
